// GCNlearnable_79748952752835
// MI455X (gfx1250) — compile-verified, matrix operand order checked by reading, not run on hardware
//
#include <hip/hip_runtime.h>
#include <hip/hip_bf16.h>
#include <hip/hip_fp16.h>

typedef __attribute__((ext_vector_type(16))) _Float16     v16h;
typedef __attribute__((ext_vector_type(8)))  float        v8f;
typedef __attribute__((ext_vector_type(4)))  unsigned int u32x4;
typedef __attribute__((ext_vector_type(8)))  int          i32x8;
typedef __attribute__((ext_vector_type(4)))  int          i32x4;

#define N_NODES 8192
#define N_BATCH 32
#define NCOL    64                     // B * OUT_F
#define SCALE_F     4096.0f
#define INV_SCALE_F (1.0f / 4096.0f)

// K tiling for the WMMA GEMM
#define KT_PER_CHUNK 8                         // 8 x K32 steps per LDS chunk (K=256)
#define CHUNK_BYTES  (KT_PER_CHUNK * 4 * 32 * 32)   // 32768 B of packed B per chunk
#define N_CHUNKS     ((N_NODES / 32) / KT_PER_CHUNK) // 32 chunks

// Workspace layout: [0,32KB) = r[8192] f32 ; [32KB, 32KB+1MB) = packed B (f16)
#define WS_R_BYTES 32768

#if defined(__gfx1250__) && __has_builtin(__builtin_amdgcn_tensor_load_to_lds) && \
    __has_builtin(__builtin_amdgcn_s_wait_tensorcnt)
#define USE_TDM 1
#else
#define USE_TDM 0
#endif

// ---------------------------------------------------------------------------
// Kernel 1: per-column online softmax stats.
// r[j] = SCALE * exp(-max_i adj[i,j]) / sum_i exp(adj[i,j] - max_i adj[i,j])
// Consecutive lanes take consecutive columns -> every row sweep is coalesced.
// ---------------------------------------------------------------------------
__global__ void gcn_colstats(const float* __restrict__ adj, float* __restrict__ r) {
  const int j = blockIdx.x * blockDim.x + threadIdx.x;   // 8192 threads
  const float* p = adj + j;
  float m = -__builtin_inff();
  float s = 0.0f;
#pragma unroll 4
  for (int i = 0; i < N_NODES; ++i) {
    float v  = p[(size_t)i * N_NODES];
    float m2 = fmaxf(m, v);
    s = s * __expf(m - m2) + __expf(v - m2);
    m = m2;
  }
  r[j] = SCALE_F * __expf(-m) / s;
}

// ---------------------------------------------------------------------------
// Kernel 2: build packed B matrix in f16, pre-swizzled into the WMMA
// B-fragment lane layout (16x16x32 f16, ISA 7.12.2):
//   lanes 0-15  hold column N=lane, K offsets {0..7, 16..23} in halfs 0..15
//   lanes 16-31 hold column N=lane-16, K offsets {8..15, 24..31}
// Bp is indexed [kt][n][lane][slot] -> each lane's 16 halfs are contiguous 32B.
// g[j,c] = (x[b*8192+j,:] @ W[:,f]) * r[j]   with c = 2*b + f
// ---------------------------------------------------------------------------
__global__ void gcn_build_b(const float* __restrict__ x, const float* __restrict__ w,
                            const float* __restrict__ r, _Float16* __restrict__ Bp) {
  const int t = blockIdx.x * blockDim.x + threadIdx.x;   // 8192 * 64 threads
  const int j = t & (N_NODES - 1);                       // fast-varying -> coalesced x/r reads
  const int c = t >> 13;                                 // 0..63
  const int b = c >> 1;
  const int f = c & 1;

  const float2 xv = ((const float2*)x)[(size_t)b * N_NODES + j];
  const float h  = xv.x * w[f] + xv.y * w[2 + f];        // W row-major (IN_F, OUT_F)
  const float g  = h * r[j];

  const int k    = j & 31;
  const int kt   = j >> 5;
  const int n    = c >> 4;
  const int lane = (c & 15) + ((k & 8) ? 16 : 0);
  const int slot = (k & 7) + ((k & 16) ? 8 : 0);
  Bp[(((size_t)(kt * 4 + n) * 32 + lane) << 4) + slot] = (_Float16)g;
}

// ---------------------------------------------------------------------------
// TDM staging: DMA one contiguous 32KB chunk of packed B into LDS.
// D# built per CDNA5 ISA 8.3-8.5 (1D tile, data_size=8B, count=1, type=2).
// 6-arg builtin form (this toolchain): (g0 u32x4, g1 i32x8, i32x4, i32x4, i32x8, cpol)
// ---------------------------------------------------------------------------
__device__ __forceinline__ unsigned lds_offset_of(const void* p) {
  // LDS aperture: flat addr low 32 bits == LDS byte offset
  return (unsigned)(unsigned long long)(uintptr_t)p;
}

#if USE_TDM
__device__ __forceinline__ void tdm_stage(const void* gsrc, unsigned ldsOff) {
  const unsigned long long ga = (unsigned long long)(uintptr_t)gsrc;
  const unsigned elems = CHUNK_BYTES / 8;    // 8-byte elements: 4096
  u32x4 g0;
  g0.x = 1u;                                     // count=1, is_restore=0, gather off
  g0.y = ldsOff;                                 // lds_addr (bytes)
  g0.z = (unsigned)ga;                           // global_addr[31:0]
  g0.w = (unsigned)(ga >> 32) | (2u << 30);      // global_addr[56:32] | type=2
  i32x8 g1;
  g1[0] = (int)(3u << 16);                       // workgroup_mask=0, data_size=3 (8B)
  g1[1] = (int)((elems & 0xFFFFu) << 16);        // tensor_dim0[15:0]
  g1[2] = (int)((elems >> 16) | (1u << 16));     // tensor_dim0[31:16], tensor_dim1=1
  g1[3] = (int)(elems << 16);                    // tile_dim0 = 4096
  g1[4] = 0;                                     // tile_dim1=0 (1D), tile_dim2=0
  g1[5] = (int)elems;                            // tensor_dim0_stride
  g1[6] = 0;
  g1[7] = 0;
  i32x4 z4 = {0, 0, 0, 0};
  i32x8 z8 = {0, 0, 0, 0, 0, 0, 0, 0};
  __builtin_amdgcn_tensor_load_to_lds(g0, g1, z4, z4, z8, 0);
}
#endif

// ---------------------------------------------------------------------------
// Kernel 3: h' = exp(adj) @ Bp via V_WMMA_F32_16X16X32_F16, f32 accumulate.
// 64 blocks x 8 waves; wave owns a 16-row M strip x all 64 N columns.
// adj read exactly once from HBM; B chunks double-buffered in LDS via TDM.
// ---------------------------------------------------------------------------
__global__ void __launch_bounds__(256)
gcn_spmm_wmma(const float* __restrict__ adj, const _Float16* __restrict__ Bp,
              float* __restrict__ out) {
  __shared__ __align__(128) unsigned char ldsB[2][CHUNK_BYTES];

  const int tid  = threadIdx.x;
  const int wave = tid >> 5;
  const int lane = tid & 31;
  const int m0   = (blockIdx.x * 8 + wave) * 16;
  const int rowA = m0 + (lane & 15);
  const int koff = (lane >> 4) << 3;                     // 0 for lanes 0-15, 8 for 16-31
  const float* arow = adj + (size_t)rowA * N_NODES;

  v8f acc[4] = {};

#if USE_TDM
  if (tid == 0) tdm_stage(Bp, lds_offset_of(&ldsB[0][0]));
#endif

  for (int ch = 0; ch < N_CHUNKS; ++ch) {
#if USE_TDM
    if (tid == 0) {
      if (ch + 1 < N_CHUNKS) {
        tdm_stage((const unsigned char*)Bp + (size_t)(ch + 1) * CHUNK_BYTES,
                  lds_offset_of(&ldsB[(ch + 1) & 1][0]));
        __builtin_amdgcn_s_wait_tensorcnt(1);  // chunk ch done, ch+1 in flight
      } else {
        __builtin_amdgcn_s_wait_tensorcnt(0);  // last chunk: drain
      }
    }
    __syncthreads();
#else
    __syncthreads();
    {
      const uint4* src = (const uint4*)((const unsigned char*)Bp + (size_t)ch * CHUNK_BYTES);
      uint4* dst = (uint4*)&ldsB[ch & 1][0];
      for (int e = tid; e < CHUNK_BYTES / 16; e += 256) dst[e] = src[e];
    }
    __syncthreads();
#endif
    const unsigned char* bbase = &ldsB[ch & 1][0];

    if (ch + 1 < N_CHUNKS)
      __builtin_prefetch(arow + (ch + 1) * (KT_PER_CHUNK * 32) + koff, 0, 1);

#pragma unroll
    for (int ktl = 0; ktl < KT_PER_CHUNK; ++ktl) {
      const int j0 = ch * (KT_PER_CHUNK * 32) + ktl * 32 + koff;
      // A fragment: row rowA, K = j0+{0..7} into halfs 0..7, j0+{16..23} into 8..15
      const float4 f0 = *(const float4*)(arow + j0);
      const float4 f1 = *(const float4*)(arow + j0 + 4);
      const float4 f2 = *(const float4*)(arow + j0 + 16);
      const float4 f3 = *(const float4*)(arow + j0 + 20);
      v16h a;
      a[0]  = (_Float16)__expf(f0.x);  a[1]  = (_Float16)__expf(f0.y);
      a[2]  = (_Float16)__expf(f0.z);  a[3]  = (_Float16)__expf(f0.w);
      a[4]  = (_Float16)__expf(f1.x);  a[5]  = (_Float16)__expf(f1.y);
      a[6]  = (_Float16)__expf(f1.z);  a[7]  = (_Float16)__expf(f1.w);
      a[8]  = (_Float16)__expf(f2.x);  a[9]  = (_Float16)__expf(f2.y);
      a[10] = (_Float16)__expf(f2.z);  a[11] = (_Float16)__expf(f2.w);
      a[12] = (_Float16)__expf(f3.x);  a[13] = (_Float16)__expf(f3.y);
      a[14] = (_Float16)__expf(f3.z);  a[15] = (_Float16)__expf(f3.w);

      const unsigned boff = ((unsigned)(ktl * 4) * 32 + (unsigned)lane) * 32;
      const v16h b0 = *(const v16h*)(bbase + boff);
      const v16h b1 = *(const v16h*)(bbase + boff + 1024);
      const v16h b2 = *(const v16h*)(bbase + boff + 2048);
      const v16h b3 = *(const v16h*)(bbase + boff + 3072);

      acc[0] = __builtin_amdgcn_wmma_f32_16x16x32_f16(false, a, false, b0, (short)0, acc[0], false, false);
      acc[1] = __builtin_amdgcn_wmma_f32_16x16x32_f16(false, a, false, b1, (short)0, acc[1], false, false);
      acc[2] = __builtin_amdgcn_wmma_f32_16x16x32_f16(false, a, false, b2, (short)0, acc[2], false, false);
      acc[3] = __builtin_amdgcn_wmma_f32_16x16x32_f16(false, a, false, b3, (short)0, acc[3], false, false);
    }
    __syncthreads();
  }

  // Epilogue: D layout -> VGPR v holds row m0+v (lanes 0-15) / m0+8+v (lanes 16-31),
  // column 16n + (lane&15). out[(b*8192 + i)*2 + f], c = 2b+f. ReLU + unscale.
  const int rbase = m0 + ((lane >> 4) << 3);
  const int lc    = lane & 15;
#pragma unroll
  for (int n = 0; n < 4; ++n) {
    const int c = 16 * n + lc;
    const int b = c >> 1;
    const int f = c & 1;
    float* op = out + (size_t)b * (N_NODES * 2) + f;
#pragma unroll
    for (int v = 0; v < 8; ++v) {
      float val = acc[n][v] * INV_SCALE_F;
      val = fmaxf(val, 0.0f);
      op[(size_t)(rbase + v) * 2] = val;
    }
  }
}

// ---------------------------------------------------------------------------
extern "C" void kernel_launch(void* const* d_in, const int* in_sizes, int n_in,
                              void* d_out, int out_size, void* d_ws, size_t ws_size,
                              hipStream_t stream) {
  (void)in_sizes; (void)n_in; (void)out_size; (void)ws_size;
  const float* x   = (const float*)d_in[0];
  // d_in[1] = batch ids (int64) -- implied by layout, unused
  const float* w   = (const float*)d_in[2];
  const float* adj = (const float*)d_in[3];
  float* out = (float*)d_out;

  float*     r  = (float*)d_ws;
  _Float16*  Bp = (_Float16*)((unsigned char*)d_ws + WS_R_BYTES);

  gcn_colstats<<<N_NODES / 256, 256, 0, stream>>>(adj, r);
  gcn_build_b<<<(N_NODES * NCOL) / 256, 256, 0, stream>>>(x, w, r, Bp);
  gcn_spmm_wmma<<<N_NODES / 128, 256, 0, stream>>>(adj, Bp, out);
}
